// TrajectoryDecoder_31653908972275
// MI455X (gfx1250) — matrix operand order repaired and checked by reading, not compiled, not run on hardware
//
#include <hip/hip_runtime.h>
#include <hip/hip_bf16.h>

// MI455X / gfx1250, wave32. bf16 WMMA 16x16x32, f32 accumulate.
typedef __attribute__((ext_vector_type(16))) __bf16 v16bf;
typedef __attribute__((ext_vector_type(8)))  float  v8f;

union BF16x16 { v16bf v; __bf16 b[16]; uint4 q[2]; };
union F32x8   { v8f v; float4 q[2]; float f[8]; };

// CDNA5 has a native TANH trans op; use it if the builtin exists.
#if __has_builtin(__builtin_amdgcn_tanhf)
__device__ __forceinline__ float tanhf_(float x)    { return __builtin_amdgcn_tanhf(x); }
__device__ __forceinline__ float sigmoidf_(float x) { return 0.5f * tanhf_(0.5f * x) + 0.5f; }
#else
__device__ __forceinline__ float sigmoidf_(float x) { return 1.f / (1.f + __expf(-x)); }
__device__ __forceinline__ float tanhf_(float x)    { return 1.f - 2.f / (1.f + __expf(2.f * x)); }
#endif

// f32 p[0..7], p[16..23] from global -> one bf16 B-fragment (one lane's 16 K-values)
__device__ __forceinline__ v16bf glb_B16(const float* p) {
  BF16x16 f;
  #pragma unroll
  for (int i = 0; i < 16; ++i) f.b[i] = (__bf16)p[(i & 7) + ((i >> 3) << 4)];   // v_cvt_pk_bf16_f32 pairs
  return f.v;
}

// LDS layout (bytes):
//   [0      , 32768 )  W_hh bf16, row-major [256][64]        (block-shared)
//   [32768  , 34816 )  W_prev f32 [256][2]                   (block-shared)
//   [34816  , 51200 )  h tiles bf16, 8 waves x [16][64]      (wave-private)
//   [51200  , 182272)  gates_static f32, 8 waves x 16 tiles x 32 lanes x 8  (wave-private)
#define LDS_BYTES 182272

__global__ __launch_bounds__(256, 1)
void traj_lstm_kernel(const float* __restrict__ context,
                      const float* __restrict__ enc,
                      const float* __restrict__ ball,
                      const float* __restrict__ Wh_ctx, const float* __restrict__ bh_ctx,
                      const float* __restrict__ Wc_ctx, const float* __restrict__ bc_ctx,
                      const float* __restrict__ W_ih,   const float* __restrict__ W_hh,
                      const float* __restrict__ b_ih,   const float* __restrict__ b_hh,
                      const float* __restrict__ W_out,  const float* __restrict__ b_out,
                      float* __restrict__ out)
{
  extern __shared__ char smem[];
  __bf16* whh   = (__bf16*)smem;
  float*  wprev = (float*)(smem + 32768);
  __bf16* htA   = (__bf16*)(smem + 34816);
  float*  gsA   = (float*)(smem + 51200);

  const int tid    = threadIdx.x;
  const int wave   = tid >> 5;
  const int lane   = tid & 31;
  const int laneLo = lane & 15;
  const bool hiH   = lane >= 16;         // lane-half selector
  const int hiOff  = hiH ? 8 : 0;        // row offset for C/D layout
  const int rowbase = blockIdx.x * 128 + wave * 16;

  __bf16* ht  = htA + wave * (16 * 64);
  float*  gsw = gsA + wave * (16 * 32 * 8);

  // ---- stage W_hh (f32 -> bf16) and W_prev (= W_ih[:, :2]) into LDS ----
  #pragma unroll 4
  for (int i = tid; i < 256 * 64; i += 256) whh[i] = (__bf16)W_hh[i];
  wprev[tid * 2 + 0] = W_ih[tid * 68 + 0];
  wprev[tid * 2 + 1] = W_ih[tid * 68 + 1];

  // ======================= Phase 0 (per wave) =======================
  // A fragments of context tile (16 x 128 -> 4 K-chunks of 32).
  // ISA A-layout: lanes 0-15 hold K = base+0..7 / base+16..23,
  //               lanes 16-31 hold K = base+8..15 / base+24..31.
  const float* ctxrow = context + (size_t)(rowbase + laneLo) * 128;
  v16bf actx[4];
  #pragma unroll
  for (int kc = 0; kc < 4; ++kc) {
    int ko = kc * 32 + (hiH ? 8 : 0);
    BF16x16 f;
    #pragma unroll
    for (int i = 0; i < 8; ++i) f.b[i]     = (__bf16)ctxrow[ko + i];
    #pragma unroll
    for (int i = 0; i < 8; ++i) f.b[8 + i] = (__bf16)ctxrow[ko + 16 + i];
    actx[kc] = f.v;
  }

  // h0 / c0 : (16x128) @ (128x64). B fragment = Wh/Wc row f32 at ko+0..7 and ko+16..23.
  // unroll 1: keep one-time phase cheap in VGPRs so the hot loop stays < 256.
  F32x8 cst[4];
  #pragma unroll 1
  for (int d = 0; d < 4; ++d) {
    int col = 16 * d + laneLo;
    F32x8 ah, ac;
    float bh = bh_ctx[col], bc = bc_ctx[col];
    #pragma unroll
    for (int r = 0; r < 8; ++r) { ah.f[r] = bh; ac.f[r] = bc; }
    #pragma unroll
    for (int kc = 0; kc < 4; ++kc) {
      int ko = kc * 32 + (hiH ? 8 : 0);
      v16bf bW = glb_B16(Wh_ctx + (size_t)col * 128 + ko);
      v16bf bC = glb_B16(Wc_ctx + (size_t)col * 128 + ko);
      ah.v = __builtin_amdgcn_wmma_f32_16x16x32_bf16(false, actx[kc], false, bW, (short)0, ah.v, false, false);
      ac.v = __builtin_amdgcn_wmma_f32_16x16x32_bf16(false, actx[kc], false, bC, (short)0, ac.v, false, false);
    }
    cst[d] = ac;                                   // c state stays in VGPRs
    #pragma unroll
    for (int r = 0; r < 8; ++r)                    // h0 -> LDS row-major bf16
      ht[(r + hiOff) * 64 + col] = (__bf16)ah.f[r];
  }

  // gates_static = enc @ W_enc.T + ball @ W_ball.T + b_ih + b_hh   (16 x 256)
  const float* encrow = enc + (size_t)(rowbase + laneLo) * 64;
  v16bf aenc[2];
  #pragma unroll
  for (int kc = 0; kc < 2; ++kc) {
    int ko = kc * 32 + (hiH ? 8 : 0);
    BF16x16 f;
    #pragma unroll
    for (int i = 0; i < 8; ++i) f.b[i]     = (__bf16)encrow[ko + i];
    #pragma unroll
    for (int i = 0; i < 8; ++i) f.b[8 + i] = (__bf16)encrow[ko + 16 + i];
    aenc[kc] = f.v;
  }
  float bx[8], by[8];
  #pragma unroll
  for (int r = 0; r < 8; ++r) {
    int row = rowbase + r + hiOff;
    bx[r] = ball[row * 2 + 0];
    by[r] = ball[row * 2 + 1];
  }
  #pragma unroll 1
  for (int j = 0; j < 16; ++j) {
    int col = 16 * j + laneLo;
    const float* wrow = W_ih + (size_t)col * 68;   // [prev(2) | ball(2) | enc(64)]
    float w2 = wrow[2], w3 = wrow[3];
    float bias = b_ih[col] + b_hh[col];
    F32x8 a;
    #pragma unroll
    for (int r = 0; r < 8; ++r) a.f[r] = bias + w2 * bx[r] + w3 * by[r];
    #pragma unroll
    for (int kc = 0; kc < 2; ++kc) {
      v16bf bfrag = glb_B16(wrow + 4 + kc * 32 + (hiH ? 8 : 0));
      a.v = __builtin_amdgcn_wmma_f32_16x16x32_bf16(false, aenc[kc], false, bfrag, (short)0, a.v, false, false);
    }
    float* gp = gsw + j * 256 + lane * 8;          // per-lane-contiguous C layout
    *(float4*)gp       = a.q[0];
    *(float4*)(gp + 4) = a.q[1];
  }

  __syncthreads();                                  // whh / wprev visible to all waves

  // Hoisted per-lane constants: output head + W_prev columns for this lane.
  float woutX[4], woutY[4];
  #pragma unroll
  for (int d = 0; d < 4; ++d) {
    woutX[d] = W_out[     16 * d + laneLo];
    woutY[d] = W_out[64 + 16 * d + laneLo];
  }
  const float b0 = b_out[0], b1 = b_out[1];

  float wp0[16], wp1[16];                           // W_prev[col(j,lane)][0..1]
  #pragma unroll
  for (int j = 0; j < 16; ++j) {
    wp0[j] = wprev[(16 * j + laneLo) * 2 + 0];
    wp1[j] = wprev[(16 * j + laneLo) * 2 + 1];
  }

  // ======================= Recurrence: 64 steps =======================
  float px[8], py[8];
  #pragma unroll
  for (int r = 0; r < 8; ++r) { px[r] = 0.f; py[r] = 0.f; }

  #pragma unroll 1
  for (int t = 0; t < 64; ++t) {
    // A fragments of current h (from LDS, per ISA A-layout)
    v16bf a0, a1;
    {
      const __bf16* base = ht + laneLo * 64 + (hiH ? 8 : 0);
      BF16x16 f;
      f.q[0] = *(const uint4*)(base + 0);
      f.q[1] = *(const uint4*)(base + 16);
      a0 = f.v;
      f.q[0] = *(const uint4*)(base + 32);
      f.q[1] = *(const uint4*)(base + 48);
      a1 = f.v;
    }

    float sx[8], sy[8];
    #pragma unroll
    for (int r = 0; r < 8; ++r) { sx[r] = 0.f; sy[r] = 0.f; }

    #pragma unroll
    for (int d = 0; d < 4; ++d) {                  // hidden 16-block
      F32x8 g4[4];
      #pragma unroll
      for (int gg = 0; gg < 4; ++gg) {             // i, f, g, o
        int j = gg * 4 + d;
        float* gp = gsw + j * 256 + lane * 8;
        F32x8 a;
        a.q[0] = *(const float4*)gp;
        a.q[1] = *(const float4*)(gp + 4);
        #pragma unroll
        for (int r = 0; r < 8; ++r) a.f[r] += wp0[j] * px[r] + wp1[j] * py[r];

        const __bf16* wb = whh + (16 * j + laneLo) * 64 + (hiH ? 8 : 0);
        BF16x16 bf_;
        bf_.q[0] = *(const uint4*)(wb + 0);
        bf_.q[1] = *(const uint4*)(wb + 16);
        a.v = __builtin_amdgcn_wmma_f32_16x16x32_bf16(false, a0, false, bf_.v, (short)0, a.v, false, false);
        bf_.q[0] = *(const uint4*)(wb + 32);
        bf_.q[1] = *(const uint4*)(wb + 48);
        a.v = __builtin_amdgcn_wmma_f32_16x16x32_bf16(false, a1, false, bf_.v, (short)0, a.v, false, false);
        g4[gg] = a;
      }
      // LSTM pointwise + new-h writeback + output-head partials
      #pragma unroll
      for (int r = 0; r < 8; ++r) {
        float ig  = sigmoidf_(g4[0].f[r]);
        float fg  = sigmoidf_(g4[1].f[r]);
        float gg_ = tanhf_(g4[2].f[r]);
        float og  = sigmoidf_(g4[3].f[r]);
        float cn  = fg * cst[d].f[r] + ig * gg_;
        cst[d].f[r] = cn;
        float hn  = og * tanhf_(cn);
        ht[(r + hiOff) * 64 + 16 * d + laneLo] = (__bf16)hn;
        sx[r] += hn * woutX[d];
        sy[r] += hn * woutY[d];
      }
    }

    // pred = h @ W_out.T + b_out : reduce over the 16 lanes of each half.
    // After the xor-tree, lanes 0-15 hold pred for rows 0-7 (VGPR r), lanes
    // 16-31 for rows 8-15 — exactly the halves the gate FMA needs next step.
    #pragma unroll
    for (int r = 0; r < 8; ++r) {
      #pragma unroll
      for (int m = 1; m < 16; m <<= 1) {
        sx[r] += __shfl_xor(sx[r], m, 32);
        sy[r] += __shfl_xor(sy[r], m, 32);
      }
      px[r] = sx[r] + b0;
      py[r] = sy[r] + b1;
    }

    if (laneLo == 0) {                             // lanes 0 and 16 emit preds
      #pragma unroll
      for (int r = 0; r < 8; ++r) {
        size_t off = (size_t)(rowbase + r + hiOff) * 128 + t * 2;
        float2 v; v.x = px[r]; v.y = py[r];
        *(float2*)(out + off) = v;
      }
    }
  }
}

extern "C" void kernel_launch(void* const* d_in, const int* in_sizes, int n_in,
                              void* d_out, int out_size, void* d_ws, size_t ws_size,
                              hipStream_t stream) {
  const float* context = (const float*)d_in[0];
  const float* enc     = (const float*)d_in[1];
  const float* ball    = (const float*)d_in[2];
  const float* Wh_ctx  = (const float*)d_in[3];
  const float* bh_ctx  = (const float*)d_in[4];
  const float* Wc_ctx  = (const float*)d_in[5];
  const float* bc_ctx  = (const float*)d_in[6];
  const float* W_ih    = (const float*)d_in[7];
  const float* W_hh    = (const float*)d_in[8];
  const float* b_ih    = (const float*)d_in[9];
  const float* b_hh    = (const float*)d_in[10];
  const float* W_out   = (const float*)d_in[11];
  const float* b_out   = (const float*)d_in[12];
  float* out = (float*)d_out;

  const int B = in_sizes[0] / 128;                 // context is (B, 128)
  dim3 grid(B / 128), block(256);                  // 8 waves x 16 rows per block
  hipLaunchKernelGGL(traj_lstm_kernel, grid, block, (size_t)LDS_BYTES, stream,
                     context, enc, ball, Wh_ctx, bh_ctx, Wc_ctx, bc_ctx,
                     W_ih, W_hh, b_ih, b_hh, W_out, b_out, out);
}
